// GLOM_74174085202112
// MI455X (gfx1250) — matrix operand order repaired and checked by reading, not compiled, not run on hardware
//
#include <hip/hip_runtime.h>
#include <hip/hip_bf16.h>

typedef __attribute__((ext_vector_type(16))) _Float16 v16h;
typedef __attribute__((ext_vector_type(8)))  _Float16 v8h;
typedef __attribute__((ext_vector_type(8)))  float    v8f;

#define WMMA_F16(A, B, C) \
    __builtin_amdgcn_wmma_f32_16x16x32_f16(false, (A), false, (B), (short)0, (C), false, false)

#define CAT16(lo, hi) \
    __builtin_shufflevector((lo), (hi), 0, 1, 2, 3, 4, 5, 6, 7, 8, 9, 10, 11, 12, 13, 14, 15)

// ---------------------------------------------------------------------------
// Prepass: per token n, compute rnorm = rsqrt(sum_c v^2 + 1e-8) and write
//   vn[n][c] = (f16)(v * rnorm)      (row-major, N x C)  -> QK operands
//   vT[c][n] = (f16) v               (transposed,  C x N) -> PV B operand
// ---------------------------------------------------------------------------
template <int C>
__global__ void glom_prep(const float* __restrict__ x,
                          _Float16* __restrict__ vn,
                          _Float16* __restrict__ vT,
                          int N) {
    const int lane = threadIdx.x & 31;
    const int wave = threadIdx.x >> 5;
    const int n    = blockIdx.x * 4 + wave;
    if (n >= N) return;

    constexpr int E = C / 32;
    float vals[E];
    float ss = 0.f;
#pragma unroll
    for (int i = 0; i < E; ++i) {
        const int c = lane + 32 * i;
        const float v = x[(size_t)c * N + n];
        vals[i] = v;
        ss += v * v;
    }
#pragma unroll
    for (int m = 1; m < 32; m <<= 1) ss += __shfl_xor(ss, m, 32);
    const float rn = rsqrtf(ss + 1e-8f);

#pragma unroll
    for (int i = 0; i < E; ++i) {
        const int c = lane + 32 * i;
        vn[(size_t)n * C + c] = (_Float16)(vals[i] * rn);
        vT[(size_t)c * N + n] = (_Float16)vals[i];
    }
}

// ---------------------------------------------------------------------------
// COLUMN-SPLIT fused attention for C=128 (levels 1 and 2, 93% of the work).
// Block = 4 waves, one 16-row M tile. Wave w independently processes column
// tiles j = w, w+4, w+8, ... end-to-end over the FULL channel dim:
//   QK (full C, 8 WMMA) -> per-lane exp*probs on its own sims -> private
//   LDS transpose of W (same-wave DS is in-order: NO barrier) -> PV (8 WMMA).
// Exactly one __syncthreads() in the kernel: the final O / rowsum merge.
// Softmax uses the constant shift M = 1/temp (cosine sims bounded by 1).
// ---------------------------------------------------------------------------
__global__ __launch_bounds__(128)
void glom_attn_cs128(const float* __restrict__ x,      // [128][N]
                     const _Float16* __restrict__ vn,  // [N][128] normalized
                     const _Float16* __restrict__ vT,  // [128][N]
                     const float* __restrict__ probs,  // [N][N]
                     float* __restrict__ out,          // [128][N]
                     int N, float invtemp, float attw) {
    constexpr int C   = 128;
    constexpr int KST = C / 32;  // 4 QK K-steps
    constexpr int TT  = C / 16;  // 8 PV accumulator tiles
    constexpr int WS  = 40;      // W staging row stride (halves): 16B aligned,
                                 // lane-half groups hit disjoint bank ranges

    __shared__ __align__(32) _Float16 wst[4][16 * WS]; // private W staging
    __shared__ float ored[4][TT][256];                 // O merge (32 KB)
    __shared__ float llds[4][16];                      // rowsum merge

    const int lane  = threadIdx.x & 31;
    const int wave  = threadIdx.x >> 5;
    const int half  = lane >> 4;
    const int l15   = lane & 15;
    const int mbase = blockIdx.x * 16;

    // Full-C persistent q fragments (A layout: M=l15, K split by lane half).
    v16h qf[KST];
    {
        const _Float16* qrow = vn + (size_t)(mbase + l15) * C + (half ? 16 : 0);
#pragma unroll
        for (int ks = 0; ks < KST; ++ks) {
            v8h lo = *(const v8h*)(qrow + 32 * ks);
            v8h hi = *(const v8h*)(qrow + 32 * ks + 8);
            qf[ks] = CAT16(lo, hi);
        }
    }

    v8f o[TT];
#pragma unroll
    for (int t = 0; t < TT; ++t) o[t] = v8f{};
    float lacc[8];
#pragma unroll
    for (int r = 0; r < 8; ++r) lacc[r] = 0.f;

    _Float16* ws = &wst[wave][0];

    for (int jb = wave * 32; jb < N; jb += 128) {
        // ---------------- QK: full-C sims for this wave's tile --------------
        v8f s0 = v8f{}, s1 = v8f{};
        {
            const _Float16* b0 = vn + (size_t)(jb + l15) * C + 16 * half;
            const _Float16* b1 = vn + (size_t)(jb + 16 + l15) * C + 16 * half;
#pragma unroll
            for (int ks = 0; ks < KST; ++ks) {
                v16h bf0 = *(const v16h*)(b0 + 32 * ks);
                v16h bf1 = *(const v16h*)(b1 + 32 * ks);
                s0 = WMMA_F16(qf[ks], bf0, s0);
                s1 = WMMA_F16(qf[ks], bf1, s1);
            }
        }

        // ------- per-lane softmax weights (constant shift M = invtemp) ------
        // C-layout: element (row = r + 8*half, col = jb + l15 [+16]).
#pragma unroll
        for (int r = 0; r < 8; ++r) {
            const int row = r + 8 * half;
            const size_t prow = (size_t)(mbase + row) * N + jb;
            if (jb + 128 < N) __builtin_prefetch(probs + prow + 128, 0, 0);
            const float p0 = __expf(s0[r] * invtemp - invtemp) * probs[prow + l15];
            const float p1 = __expf(s1[r] * invtemp - invtemp) * probs[prow + 16 + l15];
            lacc[r] += p0 + p1;
            ws[row * WS + l15]      = (_Float16)p0;
            ws[row * WS + 16 + l15] = (_Float16)p1;
        }

        // ---- gather W A-fragment (same-wave LDS: in-order, no barrier) -----
        v16h wf;
        {
            const _Float16* wr = ws + l15 * WS + (half ? 16 : 0);
            v8h lo = *(const v8h*)(wr);
            v8h hi = *(const v8h*)(wr + 8);
            wf = CAT16(lo, hi);
        }

        // ---------------- PV: o += W(16x32) @ v(32 x C) ---------------------
#pragma unroll
        for (int t = 0; t < TT; ++t) {
            const int ch = 16 * t + l15;
            v16h bf = *(const v16h*)(vT + (size_t)ch * N + jb + 16 * half);
            o[t] = WMMA_F16(wf, bf, o[t]);
        }
    }

    // ---------------- one-time merge across the 4 waves ---------------------
#pragma unroll
    for (int t = 0; t < TT; ++t)
#pragma unroll
        for (int r = 0; r < 8; ++r) ored[wave][t][r * 32 + lane] = o[t][r];
#pragma unroll
    for (int r = 0; r < 8; ++r) {
        float ts = lacc[r];
#pragma unroll
        for (int m = 1; m < 16; m <<= 1) ts += __shfl_xor(ts, m, 32);
        if (l15 == 0) llds[wave][r + 8 * half] = ts;
    }
    __syncthreads();

    const float rmix = 1.f / (1.f + attw);
    float dinv[8];
#pragma unroll
    for (int r = 0; r < 8; ++r) {
        const int row = r + 8 * half;
        const float d = llds[0][row] + llds[1][row] + llds[2][row] + llds[3][row];
        dinv[r] = 1.f / (d + 1e-8f);
    }
#pragma unroll
    for (int tt = 0; tt < 2; ++tt) {
        const int t  = wave * 2 + tt;   // this wave writes channels [wave*32, +32)
        const int ch = 16 * t + l15;
#pragma unroll
        for (int r = 0; r < 8; ++r) {
            const int token = mbase + r + 8 * half;
            const float os = ored[0][t][r * 32 + lane] + ored[1][t][r * 32 + lane] +
                             ored[2][t][r * 32 + lane] + ored[3][t][r * 32 + lane];
            const float vv = x[(size_t)ch * N + token];
            out[(size_t)ch * N + token] = (vv + attw * (os * dinv[r])) * rmix;
        }
    }
}

// ---------------------------------------------------------------------------
// Channel-split pipelined kernel for large C (levels 3 and 4).
// Pipeline (1 barrier/tile): QK(j) -> sred[j&1] -> barrier ->
//   softmax(j) -> wlds[j&1] ; PV(j-1) <- wlds[(j-1)&1]
// ---------------------------------------------------------------------------
template <int C, int WAVES>
__global__ __launch_bounds__(WAVES * 32)
void glom_attn(const float* __restrict__ x,
               const _Float16* __restrict__ vn,
               const _Float16* __restrict__ vT,
               const float* __restrict__ probs,
               float* __restrict__ out,
               int N, float invtemp, float attw) {
    constexpr int CW     = C / WAVES;
    constexpr int KSTEPS = CW / 32;
    constexpr int TT     = CW / 16;
    constexpr int RPW    = 16 / WAVES;
    constexpr int RPH    = RPW / 2;
    static_assert(CW >= 32 && RPH >= 1, "bad decomposition");

    __shared__ float sred[2][WAVES][512];
    __shared__ __align__(32) _Float16 wlds[2][16 * 32];
    __shared__ float llds[16];

    const int lane  = threadIdx.x & 31;
    const int wave  = threadIdx.x >> 5;
    const int half  = lane >> 4;
    const int l15   = lane & 15;
    const int mbase = blockIdx.x * 16;
    const int cw    = wave * CW;

    v16h qf[KSTEPS];
    {
        const _Float16* qrow = vn + (size_t)(mbase + l15) * C + cw + (half ? 16 : 0);
#pragma unroll
        for (int ks = 0; ks < KSTEPS; ++ks) {
            v8h lo = *(const v8h*)(qrow + 32 * ks);
            v8h hi = *(const v8h*)(qrow + 32 * ks + 8);
            qf[ks] = CAT16(lo, hi);
        }
    }

    v8f o[TT];
#pragma unroll
    for (int t = 0; t < TT; ++t) o[t] = v8f{};
    float lacc[RPH];
#pragma unroll
    for (int q = 0; q < RPH; ++q) lacc[q] = 0.f;

    auto do_pv = [&](int jbp, int buf) {
        const _Float16* wr = &wlds[buf][l15 * 32 + (half ? 16 : 0)];
        v8h lo = *(const v8h*)(wr);
        v8h hi = *(const v8h*)(wr + 8);
        v16h wf = CAT16(lo, hi);
#pragma unroll
        for (int t = 0; t < TT; ++t) {
            const int ch = cw + 16 * t + l15;
            v16h bf = *(const v16h*)(vT + (size_t)ch * N + jbp + 16 * half);
            o[t] = WMMA_F16(wf, bf, o[t]);
        }
    };

    const int T = N / 32;
    for (int j = 0; j < T; ++j) {
        const int jb  = j * 32;
        const int buf = j & 1;

        v8f s0 = v8f{}, s1 = v8f{};
        {
            const _Float16* b0 = vn + (size_t)(jb + l15) * C + cw + 16 * half;
            const _Float16* b1 = vn + (size_t)(jb + 16 + l15) * C + cw + 16 * half;
#pragma unroll
            for (int ks = 0; ks < KSTEPS; ++ks) {
                v16h bf0 = *(const v16h*)(b0 + 32 * ks);
                v16h bf1 = *(const v16h*)(b1 + 32 * ks);
                s0 = WMMA_F16(qf[ks], bf0, s0);
                s1 = WMMA_F16(qf[ks], bf1, s1);
            }
        }
#pragma unroll
        for (int r = 0; r < 8; ++r) {
            sred[buf][wave][r * 32 + lane]       = s0[r];
            sred[buf][wave][256 + r * 32 + lane] = s1[r];
        }

        __syncthreads();

#pragma unroll
        for (int q = 0; q < RPH; ++q) {
            const int row = wave * RPW + half * RPH + q;
            const int idx = (row & 7) * 32 + ((row >> 3) << 4) + l15;
            float a0 = 0.f, a1 = 0.f;
#pragma unroll
            for (int w = 0; w < WAVES; ++w) {
                a0 += sred[buf][w][idx];
                a1 += sred[buf][w][256 + idx];
            }
            const size_t prow = (size_t)(mbase + row) * N + jb;
            if (jb + 32 < N) __builtin_prefetch(probs + prow + 32, 0, 0);
            const float p0 = __expf(a0 * invtemp - invtemp) * probs[prow + l15];
            const float p1 = __expf(a1 * invtemp - invtemp) * probs[prow + 16 + l15];
            lacc[q] += p0 + p1;
            wlds[buf][row * 32 + l15]      = (_Float16)p0;
            wlds[buf][row * 32 + 16 + l15] = (_Float16)p1;
        }

        if (j > 0) do_pv(jb - 32, buf ^ 1);
    }

    __syncthreads();
    do_pv((T - 1) * 32, (T - 1) & 1);

#pragma unroll
    for (int q = 0; q < RPH; ++q) {
        float ts = lacc[q];
#pragma unroll
        for (int m = 1; m < 16; m <<= 1) ts += __shfl_xor(ts, m, 32);
        const int row = wave * RPW + half * RPH + q;
        if (l15 == 0) llds[row] = ts;
    }
    __syncthreads();

    const float rmix = 1.f / (1.f + attw);
    float dinv[8];
#pragma unroll
    for (int r = 0; r < 8; ++r) dinv[r] = 1.f / (llds[r + 8 * half] + 1e-8f);
#pragma unroll
    for (int t = 0; t < TT; ++t) {
        const int ch = cw + 16 * t + l15;
#pragma unroll
        for (int r = 0; r < 8; ++r) {
            const int token = mbase + r + 8 * half;
            const float vv = x[(size_t)ch * N + token];
            out[(size_t)ch * N + token] = (vv + attw * (o[t][r] * dinv[r])) * rmix;
        }
    }
}

// ---------------------------------------------------------------------------
extern "C" void kernel_launch(void* const* d_in, const int* in_sizes, int n_in,
                              void* d_out, int out_size, void* d_ws, size_t ws_size,
                              hipStream_t stream) {
    (void)in_sizes; (void)n_in; (void)out_size; (void)ws_size;

    const float* x1 = (const float*)d_in[0];   // [128][64][64]
    const float* x2 = (const float*)d_in[1];   // [128][64][64]
    const float* x3 = (const float*)d_in[2];   // [256][32][32]
    const float* x4 = (const float*)d_in[3];   // [512][16][16]
    const float* probs1 = (const float*)d_in[4];  // 4096^2
    const float* probs3 = (const float*)d_in[5];  // 1024^2
    const float* probs4 = (const float*)d_in[6];  // 256^2

    float* y1 = (float*)d_out;
    float* y2 = y1 + 128 * 4096;
    float* y3 = y2 + 128 * 4096;
    float* y4 = y3 + 256 * 1024;

    _Float16* w   = (_Float16*)d_ws;
    _Float16* vn1 = w;                    _Float16* vT1 = vn1 + 4096 * 128;
    _Float16* vn2 = vT1 + 4096 * 128;     _Float16* vT2 = vn2 + 4096 * 128;
    _Float16* vn3 = vT2 + 4096 * 128;     _Float16* vT3 = vn3 + 1024 * 256;
    _Float16* vn4 = vT3 + 1024 * 256;     _Float16* vT4 = vn4 + 256 * 512;

    glom_prep<128><<<1024, 128, 0, stream>>>(x1, vn1, vT1, 4096);
    glom_prep<128><<<1024, 128, 0, stream>>>(x2, vn2, vT2, 4096);
    glom_prep<256><<<256,  128, 0, stream>>>(x3, vn3, vT3, 1024);
    glom_prep<512><<<64,   128, 0, stream>>>(x4, vn4, vT4, 256);

    glom_attn_cs128<<<256, 128, 0, stream>>>(x1, vn1, vT1, probs1, y1, 4096,
                                             10.0f, 0.5f);
    glom_attn_cs128<<<256, 128, 0, stream>>>(x2, vn2, vT2, probs1, y2, 4096,
                                             5.0f, 1.0f);
    glom_attn<256, 8><<<64, 256, 0, stream>>>(x3, vn3, vT3, probs3, y3, 1024,
                                              (float)(1.0 / 0.3), 2.0f);
    glom_attn<512, 8><<<16, 256, 0, stream>>>(x4, vn4, vT4, probs4, y4, 256,
                                              1.0f, 2.0f);
}
